// TokenAttentiveReadout_75144747810915
// MI455X (gfx1250) — matrix operand order repaired and checked by reading, not compiled, not run on hardware
//
#include <hip/hip_runtime.h>
#include <hip/hip_fp16.h>
#include <math.h>

typedef __attribute__((ext_vector_type(16))) _Float16 v16h;
typedef __attribute__((ext_vector_type(8)))  float    v8f;

#define BATCH   256
#define NTOK    1024
#define D_TOKEN 256
#define HIDDEN  64

__device__ __forceinline__ v8f wmma_f16(v16h a, v16h b, v8f c) {
  // D = A(16x32 f16) * B(32x16 f16) + C(16x16 f32)
  return __builtin_amdgcn_wmma_f32_16x16x32_f16(
      /*neg_a=*/false, a, /*neg_b=*/false, b,
      /*c_mod=*/(short)0, c, /*reuse_a=*/false, /*reuse_b=*/false);
}

// tanh-form GELU: v/(1+exp(-2*0.7978845608*(v + 0.044715 v^3))).
// Error ~1e-3 relative — same order as the f16 quantization already applied
// to the WMMA inputs; branch-free (1 trans op) vs ~40-op divergent erff.
__device__ __forceinline__ float gelu_fast(float v) {
  float u = v * (v * v * 0.044715f + 1.0f);
  float t = __expf(u * -1.5957691216057308f);
  return __fdividef(v, 1.0f + t);
}

// ---------------------------------------------------------------------------
// Kernel 0: repack W1 [D=256][H=64] f32 -> f16 WMMA B-fragment order.
// Fragment (kb, t): 32 lanes x 16 f16.  Lane L: n = L%16, Kbase = (L<16)?0:8,
// element i <-> K = kb*32 + Kbase + (i%8) + 16*(i/8), column h = t*16 + n.
// ---------------------------------------------------------------------------
__global__ __launch_bounds__(256)
void pack_w1(const float* __restrict__ W1, _Float16* __restrict__ out) {
  int idx = blockIdx.x * blockDim.x + threadIdx.x;   // 0 .. 8*4*32*16-1
  if (idx >= 8 * 4 * 32 * 16) return;
  int i    = idx & 15;
  int lane = (idx >> 4) & 31;
  int t    = (idx >> 9) & 3;
  int kb   = idx >> 11;
  int kbase = (lane < 16) ? 0 : 8;
  int k = kb * 32 + kbase + (i & 7) + ((i >> 3) << 4);
  int h = t * 16 + (lane & 15);
  out[idx] = (_Float16)W1[k * HIDDEN + h];
}

// ---------------------------------------------------------------------------
// Kernel 1: per-token LayerNorm -> (xn @ W1 + b1) -> GELU -> dot W2 = logits.
// One wave handles 16 tokens.  blockDim = 128 (4 independent waves).
// ---------------------------------------------------------------------------
__global__ __launch_bounds__(128)
void logits_kernel(const float* __restrict__ x,
                   const float* __restrict__ gamma,
                   const float* __restrict__ beta,
                   const _Float16* __restrict__ w1p,
                   const float* __restrict__ b1,
                   const float* __restrict__ w2,
                   float* __restrict__ logits) {
  const int lane = threadIdx.x & 31;
  const int wave = threadIdx.x >> 5;
  const int grp  = blockIdx.x * 4 + wave;     // 16-token group, 0..16383
  const int b    = grp >> 6;                  // batch index
  const int n0   = (grp & 63) << 4;           // first token of this tile
  const int M    = lane & 15;                 // A-matrix row (token) for this lane
  const int kbaseA = (lane < 16) ? 0 : 8;     // A fragment K base within 32-block

  const float* xrow = x + ((size_t)b * NTOK + (n0 + M)) * D_TOKEN;

  // ---- pass A: LayerNorm statistics over this lane's 128 K values ----
  float s = 0.f, ss = 0.f;
  auto acc4 = [&](float4 q) {
    s  += q.x + q.y + q.z + q.w;
    ss += q.x * q.x + q.y * q.y + q.z * q.z + q.w * q.w;
  };
  #pragma unroll
  for (int kb = 0; kb < 8; ++kb) {
    const float* p0 = xrow + kb * 32 + kbaseA;
    acc4(*(const float4*)(p0));
    acc4(*(const float4*)(p0 + 4));
    acc4(*(const float4*)(p0 + 16));
    acc4(*(const float4*)(p0 + 20));
  }
  s  += __shfl_xor(s, 16);     // combine complementary K halves (lane ^ 16)
  ss += __shfl_xor(ss, 16);
  const float mu   = s * (1.f / 256.f);
  const float rstd = rsqrtf(ss * (1.f / 256.f) - mu * mu + 1e-5f);

  // ---- pass B: normalize, pack A fragments, WMMA into 4 N-tiles ----
  v8f acc0 = {}, acc1 = {}, acc2 = {}, acc3 = {};
  for (int kb = 0; kb < 8; ++kb) {
    const float* p0 = xrow  + kb * 32 + kbaseA;
    const float* g0 = gamma + kb * 32 + kbaseA;
    const float* e0 = beta  + kb * 32 + kbaseA;
    float4 a0 = *(const float4*)(p0);       float4 a1 = *(const float4*)(p0 + 4);
    float4 a2 = *(const float4*)(p0 + 16);  float4 a3 = *(const float4*)(p0 + 20);
    float4 g0v = *(const float4*)(g0);      float4 g1v = *(const float4*)(g0 + 4);
    float4 g2v = *(const float4*)(g0 + 16); float4 g3v = *(const float4*)(g0 + 20);
    float4 e0v = *(const float4*)(e0);      float4 e1v = *(const float4*)(e0 + 4);
    float4 e2v = *(const float4*)(e0 + 16); float4 e3v = *(const float4*)(e0 + 20);

    float r0[8] = {a0.x, a0.y, a0.z, a0.w, a1.x, a1.y, a1.z, a1.w};
    float r1[8] = {a2.x, a2.y, a2.z, a2.w, a3.x, a3.y, a3.z, a3.w};
    float gg0[8] = {g0v.x, g0v.y, g0v.z, g0v.w, g1v.x, g1v.y, g1v.z, g1v.w};
    float gg1[8] = {g2v.x, g2v.y, g2v.z, g2v.w, g3v.x, g3v.y, g3v.z, g3v.w};
    float ee0[8] = {e0v.x, e0v.y, e0v.z, e0v.w, e1v.x, e1v.y, e1v.z, e1v.w};
    float ee1[8] = {e2v.x, e2v.y, e2v.z, e2v.w, e3v.x, e3v.y, e3v.z, e3v.w};

    v16h a;
    #pragma unroll
    for (int i = 0; i < 8; ++i) {
      a[i]     = (_Float16)((r0[i] - mu) * rstd * gg0[i] + ee0[i]); // K = Kb+i
      a[i + 8] = (_Float16)((r1[i] - mu) * rstd * gg1[i] + ee1[i]); // K = Kb+i+16
    }

    const v16h* bf = ((const v16h*)w1p) + (size_t)(kb * 4) * 32 + lane;
    acc0 = wmma_f16(a, bf[0],      acc0);
    acc1 = wmma_f16(a, bf[32],     acc1);
    acc2 = wmma_f16(a, bf[64],     acc2);
    acc3 = wmma_f16(a, bf[96],     acc3);
  }

  // ---- epilogue: +b1, GELU, dot with W2, reduce across lanes ----
  // C/D layout: lane L column = L%16, row r -> token (r + (L<16 ? 0 : 8)).
  float p[8] = {0.f, 0.f, 0.f, 0.f, 0.f, 0.f, 0.f, 0.f};
  auto epi = [&](const v8f& A, int t) {
    float b1v = b1[t * 16 + M];
    float w2v = w2[t * 16 + M];
    #pragma unroll
    for (int r = 0; r < 8; ++r) {
      float v = A[r] + b1v;
      p[r] = fmaf(gelu_fast(v), w2v, p[r]);
    }
  };
  epi(acc0, 0); epi(acc1, 1); epi(acc2, 2); epi(acc3, 3);

  #pragma unroll
  for (int r = 0; r < 8; ++r) {           // sum over the 16 columns of each half
    p[r] += __shfl_xor(p[r], 1);
    p[r] += __shfl_xor(p[r], 2);
    p[r] += __shfl_xor(p[r], 4);
    p[r] += __shfl_xor(p[r], 8);
  }
  if (M == 0) {                            // lanes 0 and 16 hold the row sums
    int mbase = (lane < 16) ? 0 : 8;
    float* lp = logits + (size_t)b * NTOK + n0 + mbase;
    #pragma unroll
    for (int r = 0; r < 8; ++r) lp[r] = p[r];
  }
}

// ---------------------------------------------------------------------------
// Kernel 2: softmax over N per batch row.  256 threads, 4 logits each.
// ---------------------------------------------------------------------------
__global__ __launch_bounds__(256)
void softmax_kernel(const float* __restrict__ logits, float* __restrict__ weights) {
  __shared__ float red[8];
  const int b  = blockIdx.x;
  const int tid = threadIdx.x;
  const int wid = tid >> 5, ln = tid & 31;
  float4 v = *(const float4*)(logits + (size_t)b * NTOK + tid * 4);

  float m = fmaxf(fmaxf(v.x, v.y), fmaxf(v.z, v.w));
  #pragma unroll
  for (int off = 1; off < 32; off <<= 1) m = fmaxf(m, __shfl_xor(m, off));
  if (ln == 0) red[wid] = m;
  __syncthreads();
  if (wid == 0) {
    float t = (ln < 8) ? red[ln] : -3.4e38f;
    #pragma unroll
    for (int off = 1; off < 8; off <<= 1) t = fmaxf(t, __shfl_xor(t, off));
    if (ln == 0) red[0] = t;
  }
  __syncthreads();
  m = red[0];
  __syncthreads();

  float4 e;
  e.x = __expf(v.x - m); e.y = __expf(v.y - m);
  e.z = __expf(v.z - m); e.w = __expf(v.w - m);
  float s = e.x + e.y + e.z + e.w;
  #pragma unroll
  for (int off = 1; off < 32; off <<= 1) s += __shfl_xor(s, off);
  if (ln == 0) red[wid] = s;
  __syncthreads();
  if (wid == 0) {
    float t = (ln < 8) ? red[ln] : 0.f;
    #pragma unroll
    for (int off = 1; off < 8; off <<= 1) t += __shfl_xor(t, off);
    if (ln == 0) red[0] = t;
  }
  __syncthreads();
  float inv = 1.f / red[0];
  float4 w; w.x = e.x * inv; w.y = e.y * inv; w.z = e.z * inv; w.w = e.w * inv;
  *(float4*)(weights + (size_t)b * NTOK + tid * 4) = w;
}

// ---------------------------------------------------------------------------
// Kernel 3: summary[b,d] = sum_n weights[b,n] * x[b,n,d].  256 threads = D.
// ---------------------------------------------------------------------------
__global__ __launch_bounds__(256)
void summary_kernel(const float* __restrict__ x, const float* __restrict__ weights,
                    float* __restrict__ summary) {
  __shared__ float wl[NTOK];
  const int b = blockIdx.x;
  const int tid = threadIdx.x;
  *(float4*)(wl + tid * 4) = *(const float4*)(weights + (size_t)b * NTOK + tid * 4);
  __syncthreads();

  const float* xb = x + (size_t)b * NTOK * D_TOKEN + tid;
  float a0 = 0.f, a1 = 0.f, a2 = 0.f, a3 = 0.f;
  for (int n = 0; n < NTOK; n += 4) {
    __builtin_prefetch(xb + (size_t)(n + 8) * D_TOKEN, 0, 0); // global_prefetch_b8
    a0 = fmaf(wl[n + 0], xb[(size_t)(n + 0) * D_TOKEN], a0);
    a1 = fmaf(wl[n + 1], xb[(size_t)(n + 1) * D_TOKEN], a1);
    a2 = fmaf(wl[n + 2], xb[(size_t)(n + 2) * D_TOKEN], a2);
    a3 = fmaf(wl[n + 3], xb[(size_t)(n + 3) * D_TOKEN], a3);
  }
  summary[(size_t)b * D_TOKEN + tid] = (a0 + a1) + (a2 + a3);
}

// ---------------------------------------------------------------------------
extern "C" void kernel_launch(void* const* d_in, const int* in_sizes, int n_in,
                              void* d_out, int out_size, void* d_ws, size_t ws_size,
                              hipStream_t stream) {
  const float* x     = (const float*)d_in[0];  // [B,N,D]
  const float* gamma = (const float*)d_in[1];  // [D]
  const float* beta  = (const float*)d_in[2];  // [D]
  const float* W1    = (const float*)d_in[3];  // [D,H]
  const float* b1    = (const float*)d_in[4];  // [H]
  const float* W2    = (const float*)d_in[5];  // [H]
  // d_in[6] = b2: constant shift cancels in softmax; weights/summary unaffected.

  _Float16* w1p   = (_Float16*)d_ws;                       // 32 KB fragment pack
  float*    logits = (float*)((char*)d_ws + 64 * 1024);    // 1 MB [B,N]
  float*    summary = (float*)d_out;                       // [B,D]
  float*    weights = (float*)d_out + BATCH * D_TOKEN;     // [B,N]

  pack_w1<<<64, 256, 0, stream>>>(W1, w1p);
  logits_kernel<<<(BATCH * NTOK / 16) / 4, 128, 0, stream>>>(
      x, gamma, beta, w1p, b1, W2, logits);
  softmax_kernel<<<BATCH, 256, 0, stream>>>(logits, weights);
  summary_kernel<<<BATCH, 256, 0, stream>>>(x, weights, summary);
}
